// LT_MultiHeadAttention_39900246180056
// MI455X (gfx1250) — matrix operand order repaired and checked by reading, not compiled, not run on hardware
//
#include <hip/hip_runtime.h>
#include <hip/hip_bf16.h>

// ---------------------------------------------------------------------------
// LT_MultiHeadAttention for MI455X (gfx1250): f16 WMMA + async-to-LDS pipeline
//   B=2, N=2048, E=512, H=8, d=64
//   1) convert x/xT/xh f32->f16 ; convert+transpose 8 weights
//   2) 7 projection GEMMs: double-buffered LDS tiles filled with
//      global_load_async_to_lds_b128 (ASYNCcnt), 4 WMMAs / wave / K-step
//      k-projections stored per-head transposed Kt[bh][d][n]
//   3) flash attention: block = 4 waves x 16 q-rows of one (b,h); K/V chunks
//      staged cooperatively via async-to-LDS (double buffered); gamma factor
//      only evaluated in the diagonal band; P routed through LDS to convert
//      WMMA C-layout -> A-layout for the PV WMMAs
//   4) output projection GEMM -> f32
// ---------------------------------------------------------------------------

typedef __attribute__((ext_vector_type(16))) _Float16 v16h;
typedef __attribute__((ext_vector_type(8)))  _Float16 v8h;
typedef __attribute__((ext_vector_type(8)))  float    v8f;

#define EDIM 512
#define NSEQ 2048
#define NHEAD 8
#define HDIM 64
#define NTOK 4096            // B*N
#define TOKE (NTOK * EDIM)   // elements per (B,N,E) tensor
#define WELE (EDIM * EDIM)

static __device__ __forceinline__ v8f wmma_f16(v16h a, v16h b, v8f c) {
  return __builtin_amdgcn_wmma_f32_16x16x32_f16(
      false, a, false, b, (short)0, c, false, false);
}

// 16x32 f16 A-fragment: lane L -> row M=L%16; halfs[0:8)=K[g*8..],
// halfs[8:16)=K[16+g*8..], g=L/16 (ISA 7.12.2 16-bit A layout).
static __device__ __forceinline__ v16h frag16(const _Float16* lo, const _Float16* hi) {
  v8h a = *(const v8h*)lo;
  v8h b = *(const v8h*)hi;
  v16h r;
#pragma unroll
  for (int i = 0; i < 8; ++i) { r[i] = a[i]; r[i + 8] = b[i]; }
  return r;
}

// gfx1250 async copy: 16B global -> LDS, tracked by ASYNCcnt.
static __device__ __forceinline__ void async_ld_b128(unsigned lds, const void* g) {
  asm volatile("global_load_async_to_lds_b128 %0, %1, off"
               :: "v"(lds), "v"((unsigned long long)(size_t)g) : "memory");
}
static __device__ __forceinline__ void wait_async0() {
  asm volatile("s_wait_asynccnt 0x0" ::: "memory");
}
static __device__ __forceinline__ unsigned ldsoff(const void* p) {
  return (unsigned)(size_t)p;   // generic LDS pointer low 32 bits = LDS address
}

// --------------------------- conversion kernels ----------------------------

__global__ void cvt_f32_f16(const float* __restrict__ src, _Float16* __restrict__ dst, int n) {
  int i = blockIdx.x * blockDim.x + threadIdx.x;
  if (i < n) dst[i] = (_Float16)src[i];
}

// Wt[k*E + n] = (f16) W[n*E + k]
__global__ void cvt_w_transpose(const float* __restrict__ W, _Float16* __restrict__ Wt) {
  int i = blockIdx.x * blockDim.x + threadIdx.x;   // i = k*E + n
  int k = i >> 9, n = i & (EDIM - 1);
  Wt[i] = (_Float16)W[n * EDIM + k];
}

// ------------------------------ projection GEMM ----------------------------
// Y[M x 512] = A[M x 512] @ Bw[512 x 512] + bias
// mode 0: f16 row-major; mode 1: f16 per-head transposed Kt[bh][d][n];
// mode 2: f32 row-major (final output)

__global__ __launch_bounds__(128) void gemm_proj(
    const _Float16* __restrict__ A, const _Float16* __restrict__ Bw,
    const float* __restrict__ bias, _Float16* __restrict__ out16,
    float* __restrict__ out32, int mode)
{
  __shared__ __align__(16) _Float16 sA[2][64][32];
  __shared__ __align__(16) _Float16 sB[2][32][64];

  const int tid  = threadIdx.x;
  const int wave = tid >> 5;
  const int lane = tid & 31;
  const int g    = lane >> 4;
  const int lm   = lane & 15;
  const int m0   = blockIdx.x * 64;
  const int n0   = blockIdx.y * 64;

  const int rowA = tid >> 1, cbA = (tid & 1) * 16;   // A tile 64x32
  const int rowB = tid >> 2, cbB = (tid & 3) * 16;   // B tile 32x64

  auto stage = [&](int buf, int k0) {
    const _Float16* sa = &A[(size_t)(m0 + rowA) * EDIM + k0 + cbA];
    async_ld_b128(ldsoff(&sA[buf][rowA][cbA]),     sa);
    async_ld_b128(ldsoff(&sA[buf][rowA][cbA + 8]), sa + 8);
    const _Float16* sb = &Bw[(size_t)(k0 + rowB) * EDIM + n0 + cbB];
    async_ld_b128(ldsoff(&sB[buf][rowB][cbB]),     sb);
    async_ld_b128(ldsoff(&sB[buf][rowB][cbB + 8]), sb + 8);
  };

  v8f acc[4] = {};

  stage(0, 0);
  wait_async0();
  __syncthreads();

  int p = 0;
  for (int k0 = 0; k0 < EDIM; k0 += 32, p ^= 1) {
    if (k0 + 32 < EDIM) stage(p ^ 1, k0 + 32);   // prefetch next tile (async)

    const int arow = wave * 16 + lm;
    v16h af = frag16(&sA[p][arow][g * 8], &sA[p][arow][16 + g * 8]);
#pragma unroll
    for (int c = 0; c < 4; ++c) {
      v16h bf;
#pragma unroll
      for (int i = 0; i < 8; ++i) { bf[i] = sB[p][g * 8 + i][c * 16 + lm]; bf[i + 8] = sB[p][16 + g * 8 + i][c * 16 + lm]; }
      acc[c] = wmma_f16(af, bf, acc[c]);
    }

    wait_async0();        // own async writes to buf p^1 visible in LDS
    __syncthreads();      // publish to all waves / retire reads of buf p
  }

#pragma unroll
  for (int c = 0; c < 4; ++c) {
#pragma unroll
    for (int r = 0; r < 8; ++r) {
      int mrow = m0 + wave * 16 + r + 8 * g;
      int ncol = n0 + c * 16 + lm;
      float v = acc[c][r] + bias[ncol];
      if (mode == 0) {
        out16[(size_t)mrow * EDIM + ncol] = (_Float16)v;
      } else if (mode == 1) {
        int b = mrow >> 11, nq = mrow & (NSEQ - 1);
        int h = ncol >> 6,  dd = ncol & (HDIM - 1);
        out16[((size_t)((b << 3) + h) * HDIM + dd) * NSEQ + nq] = (_Float16)v;
      } else {
        out32[(size_t)mrow * EDIM + ncol] = v;
      }
    }
  }
}

// ------------------------------ flash attention ----------------------------
// Block: 4 waves = 64 q-rows of one (b,h). Per 32-key chunk: cooperative
// async staging of 3 K-mats (64d x 32k each) + V (32k x 64d) into LDS
// (double buffered), 12 WMMAs for summed QK^T, banded-gamma online softmax,
// P through per-wave LDS slice, 4 WMMAs for PV.

__global__ __launch_bounds__(128) void flash_attn(
    const _Float16* __restrict__ Qx, const _Float16* __restrict__ Qh,
    const _Float16* __restrict__ Qt,
    const _Float16* __restrict__ Ktx, const _Float16* __restrict__ Kth,
    const _Float16* __restrict__ Ktt,
    const _Float16* __restrict__ V, _Float16* __restrict__ AO)
{
  __shared__ __align__(16) _Float16 sK[2][3][64][32];   // [buf][mat][d][key]
  __shared__ __align__(16) _Float16 sV[2][32][64];      // [buf][key][d]
  __shared__ __align__(16) _Float16 sP[4][16][32];      // per-wave P tile

  const int tid  = threadIdx.x;
  const int wave = tid >> 5;
  const int lane = tid & 31;
  const int g    = lane >> 4;
  const int lm   = lane & 15;

  const int bh = blockIdx.y;               // b*8 + h
  const int b  = bh >> 3;
  const int h  = bh & 7;
  const int q0 = blockIdx.x * 64 + wave * 16;

  const size_t rowbase = (size_t)b * NSEQ * EDIM;
  const size_t ktbase  = (size_t)bh * HDIM * NSEQ;

  // Q fragments: [matrix][d-slice of 32] (direct from global, loaded once)
  const _Float16* Qp[3] = {Qx, Qh, Qt};
  v16h qa[3][2];
#pragma unroll
  for (int m = 0; m < 3; ++m)
#pragma unroll
    for (int s = 0; s < 2; ++s) {
      const _Float16* base = Qp[m] + rowbase + (size_t)(q0 + lm) * EDIM + h * HDIM + s * 32;
      qa[m][s] = frag16(base + g * 8, base + 16 + g * 8);
    }

  const _Float16* Ktp[3] = {Ktx, Kth, Ktt};

  // cooperative async staging of one 32-key chunk (12KB K + 4KB V)
  auto stageKV = [&](int buf, int j) {
#pragma unroll
    for (int i = 0; i < 6; ++i) {          // 768 B128 for K
      int id  = tid + 128 * i;
      int row = id >> 2, seg = id & 3;     // row < 192
      int m = row >> 6, dd = row & 63;
      async_ld_b128(ldsoff(&sK[buf][m][dd][seg * 8]),
                    Ktp[m] + ktbase + (size_t)dd * NSEQ + j + seg * 8);
    }
#pragma unroll
    for (int i = 0; i < 2; ++i) {          // 256 B128 for V
      int id  = tid + 128 * i;
      int row = id >> 3, seg = id & 7;
      async_ld_b128(ldsoff(&sV[buf][row][seg * 8]),
                    V + rowbase + (size_t)(j + row) * EDIM + h * HDIM + seg * 8);
    }
  };

  float mrow[8], lrow[8];
  v8f o[4] = {};
#pragma unroll
  for (int r = 0; r < 8; ++r) { mrow[r] = -1e30f; lrow[r] = 0.0f; }

  const float scale = 0.014731391274719738f;   // 1 / (3 * sqrt(512))

  stageKV(0, 0);
  wait_async0();
  __syncthreads();

  int p = 0;
  for (int j = 0; j < NSEQ; j += 32, p ^= 1) {
    if (j + 32 < NSEQ) stageKV(p ^ 1, j + 32);   // async prefetch next chunk

    v8f s0 = {}, s1 = {};
#pragma unroll
    for (int m = 0; m < 3; ++m) {
#pragma unroll
      for (int s = 0; s < 2; ++s) {
        v16h b0, b1;
#pragma unroll
        for (int i = 0; i < 8; ++i) {
          int d0 = s * 32 + g * 8 + i, d1 = d0 + 16;
          b0[i] = sK[p][m][d0][lm];      b0[i + 8] = sK[p][m][d1][lm];
          b1[i] = sK[p][m][d0][16 + lm]; b1[i + 8] = sK[p][m][d1][16 + lm];
        }
        s0 = wmma_f16(qa[m][s], b0, s0);
        s1 = wmma_f16(qa[m][s], b1, s1);
      }
    }

    // gamma differs from 1 by <3e-13 outside |k-q| <= 12 band: uniform branch
    const bool need_gamma = (j + 31 > q0) && (j <= q0 + 27);

    float p0[8], p1[8];
#pragma unroll
    for (int r = 0; r < 8; ++r) {
      int qp = q0 + 8 * g + r;
      float g0 = 1.0f, g1 = 1.0f;
      if (need_gamma) {
        float d0 = (float)(j + lm - qp);
        float d1 = d0 + 16.0f;
        g0 = 1.0f + ((d0 > 0.0f) ? __expf(-d0 * d0 * 0.2f) : 0.0f);
        g1 = 1.0f + ((d1 > 0.0f) ? __expf(-d1 * d1 * 0.2f) : 0.0f);
      }
      float l0 = s0[r] * scale * g0;
      float l1 = s1[r] * scale * g1;
      float mx = fmaxf(l0, l1);
      for (int msk = 1; msk < 16; msk <<= 1) mx = fmaxf(mx, __shfl_xor(mx, msk, 32));
      float mnew = fmaxf(mrow[r], mx);
      float a0 = __expf(l0 - mnew);
      float a1 = __expf(l1 - mnew);
      float sm = a0 + a1;
      for (int msk = 1; msk < 16; msk <<= 1) sm += __shfl_xor(sm, msk, 32);
      float alpha = __expf(mrow[r] - mnew);
      lrow[r] = lrow[r] * alpha + sm;
      mrow[r] = mnew;
      p0[r] = a0; p1[r] = a1;
#pragma unroll
      for (int c = 0; c < 4; ++c) o[c][r] *= alpha;
    }

    // C-layout -> per-wave LDS slice (DS ops are in-order within a wave)
#pragma unroll
    for (int r = 0; r < 8; ++r) {
      sP[wave][r + 8 * g][lm]      = (_Float16)p0[r];
      sP[wave][r + 8 * g][16 + lm] = (_Float16)p1[r];
    }

    // PV: A-frag of P (16q x 32k), B-frags of V (32k x 16d) from LDS
    v16h pa = frag16(&sP[wave][lm][g * 8], &sP[wave][lm][16 + g * 8]);
#pragma unroll
    for (int c = 0; c < 4; ++c) {
      v16h bf;
#pragma unroll
      for (int i = 0; i < 8; ++i) { bf[i] = sV[p][g * 8 + i][c * 16 + lm]; bf[i + 8] = sV[p][16 + g * 8 + i][c * 16 + lm]; }
      o[c] = wmma_f16(pa, bf, o[c]);
    }

    wait_async0();
    __syncthreads();
  }

#pragma unroll
  for (int r = 0; r < 8; ++r) {
    float inv = __builtin_amdgcn_rcpf(lrow[r]);
#pragma unroll
    for (int c = 0; c < 4; ++c) {
      float v = o[c][r] * inv;
      AO[rowbase + (size_t)(q0 + r + 8 * g) * EDIM + h * HDIM + c * 16 + lm] = (_Float16)v;
    }
  }
}

// ------------------------------- launch ------------------------------------

extern "C" void kernel_launch(void* const* d_in, const int* in_sizes, int n_in,
                              void* d_out, int out_size, void* d_ws, size_t ws_size,
                              hipStream_t stream) {
  (void)in_sizes; (void)n_in; (void)out_size; (void)ws_size;

  const float* x   = (const float*)d_in[0];
  const float* xT  = (const float*)d_in[1];
  const float* xh  = (const float*)d_in[2];
  const float* W_[8]  = {(const float*)d_in[3],  (const float*)d_in[5],
                         (const float*)d_in[7],  (const float*)d_in[9],
                         (const float*)d_in[11], (const float*)d_in[13],
                         (const float*)d_in[15], (const float*)d_in[17]};
  const float* b_[8]  = {(const float*)d_in[4],  (const float*)d_in[6],
                         (const float*)d_in[8],  (const float*)d_in[10],
                         (const float*)d_in[12], (const float*)d_in[14],
                         (const float*)d_in[16], (const float*)d_in[18]};
  // weight order: qx, kx, vx, qh, kh, qt, kt, p
  float* out = (float*)d_out;

  _Float16* w16  = (_Float16*)d_ws;            // ~48 MB workspace
  _Float16* X16x = w16;
  _Float16* X16T = X16x + TOKE;
  _Float16* X16h = X16T + TOKE;
  _Float16* WT0  = X16h + TOKE;                // 8 transposed f16 weights
  _Float16* Q16x = WT0 + (size_t)8 * WELE;
  _Float16* Q16h = Q16x + TOKE;
  _Float16* Q16t = Q16h + TOKE;
  _Float16* Kt16x = Q16t + TOKE;
  _Float16* Kt16h = Kt16x + TOKE;
  _Float16* Kt16t = Kt16h + TOKE;
  _Float16* V16   = Kt16t + TOKE;
  _Float16* AO16  = V16 + TOKE;

  cvt_f32_f16<<<TOKE / 256, 256, 0, stream>>>(x,  X16x, TOKE);
  cvt_f32_f16<<<TOKE / 256, 256, 0, stream>>>(xT, X16T, TOKE);
  cvt_f32_f16<<<TOKE / 256, 256, 0, stream>>>(xh, X16h, TOKE);

  for (int i = 0; i < 8; ++i)
    cvt_w_transpose<<<WELE / 256, 256, 0, stream>>>(W_[i], WT0 + (size_t)i * WELE);

  dim3 gGemm(NTOK / 64, EDIM / 64);

  gemm_proj<<<gGemm, 128, 0, stream>>>(X16x, WT0 + 0 * (size_t)WELE, b_[0], Q16x,  nullptr, 0); // qx
  gemm_proj<<<gGemm, 128, 0, stream>>>(X16x, WT0 + 1 * (size_t)WELE, b_[1], Kt16x, nullptr, 1); // kx
  gemm_proj<<<gGemm, 128, 0, stream>>>(X16x, WT0 + 2 * (size_t)WELE, b_[2], V16,   nullptr, 0); // vx
  gemm_proj<<<gGemm, 128, 0, stream>>>(X16h, WT0 + 3 * (size_t)WELE, b_[3], Q16h,  nullptr, 0); // qh
  gemm_proj<<<gGemm, 128, 0, stream>>>(X16h, WT0 + 4 * (size_t)WELE, b_[4], Kt16h, nullptr, 1); // kh
  gemm_proj<<<gGemm, 128, 0, stream>>>(X16T, WT0 + 5 * (size_t)WELE, b_[5], Q16t,  nullptr, 0); // qt
  gemm_proj<<<gGemm, 128, 0, stream>>>(X16T, WT0 + 6 * (size_t)WELE, b_[6], Kt16t, nullptr, 1); // kt

  dim3 gAttn(NSEQ / 64, 16);   // (q-blocks, b*h)
  flash_attn<<<gAttn, 128, 0, stream>>>(Q16x, Q16h, Q16t, Kt16x, Kt16h, Kt16t, V16, AO16);

  gemm_proj<<<gGemm, 128, 0, stream>>>(AO16, WT0 + 7 * (size_t)WELE, b_[7], nullptr, out, 2);
}
